// RNN_24215025615335
// MI455X (gfx1250) — compile-verified, matrix operand order checked by reading, not run on hardware
//
#include <hip/hip_runtime.h>

// ---------------------------------------------------------------------------
// Persistent LSTM recurrence for MI455X (gfx1250, wave32, WMMA).
//   T=16384, B=32, H=96, G=4H=384.
//   grid = 2 blocks (16 batch rows each), block = 256 threads = 8 wave32.
//   Per step:  g(16x384) = h(16x96,f16) @ w_hh^T(96x384,f16)  via
//              v_wmma_f32_16x16x32_f16 (24 N-tiles x 3 K-frags, 9/wave),
//              then activations (v_tanh_f32) + c/h update in registers,
//              then out[t,b] = h . fc_w + fc_b + x[t,b].
//   w_hh fragments are loop-invariant -> held in VGPRs for all 16384 steps.
// ---------------------------------------------------------------------------

typedef __attribute__((ext_vector_type(16))) _Float16 v16h;
typedef __attribute__((ext_vector_type(8)))  _Float16 v8h;
typedef __attribute__((ext_vector_type(8)))  float    v8f;

#define T_STEPS  16384
#define BATCH    32
#define HDIM     96
#define GDIM     384          // 4*H
#define BPB      16           // batch rows per block (one WMMA M-tile)
#define NTHREADS 256          // 8 wave32
#define TPW      3            // N-tiles per wave (24 tiles / 8 waves)
#define KFRAGS   3            // 96 / 32
#define CHUNK    64           // x staging chunk (timesteps)
#define EPT      6            // elementwise elems/thread: 16*96/256

// LDS row strides (bank-conflict tuned, 64 banks x 4B)
#define HS   104              // h16 stride in halves: 16B lane windows tile banks
#define H32S 97               // h32 stride in floats: odd -> conflict-free dot
#define GS   388              // gate stride in floats: 8*GS % 64 != 0

__device__ __forceinline__ float tanh_fast(float v) {
    float r;
    // CDNA5 TRANS op; v_nop satisfies the 1-instruction TRANS result hazard.
    asm volatile("v_tanh_f32 %0, %1\n\tv_nop" : "=v"(r) : "v"(v));
    return r;
}
__device__ __forceinline__ float sigmoid_fast(float v) {
    return __builtin_fmaf(0.5f, tanh_fast(0.5f * v), 0.5f);
}

__global__ __launch_bounds__(NTHREADS, 1)
void lstm_persistent_kernel(const float* __restrict__ x,
                            const float* __restrict__ w_ih,
                            const float* __restrict__ w_hh,
                            const float* __restrict__ b_ih,
                            const float* __restrict__ b_hh,
                            const float* __restrict__ fc_w,
                            const float* __restrict__ fc_b,
                            float* __restrict__ out)
{
    __shared__ __align__(16) _Float16 h16[BPB * HS];      // h_t as f16 (WMMA A source)
    __shared__ __align__(16) float    h32[BPB * H32S];    // h_t as f32 (output dot)
    __shared__ __align__(16) float    g_lds[BPB * GS];    // WMMA D: raw gates
    __shared__ __align__(16) float    x_lds[CHUNK * BPB]; // staged input chunk
    __shared__ __align__(16) float    fc_lds[HDIM];

    const int tid  = threadIdx.x;
    const int lane = tid & 31;
    const int wave = tid >> 5;
    const int b0   = blockIdx.x * BPB;    // batch offset of this block

    // ---------------- one-time init ----------------
    for (int i = tid; i < BPB * HS;   i += NTHREADS) h16[i] = (_Float16)0.f;
    for (int i = tid; i < BPB * H32S; i += NTHREADS) h32[i] = 0.f;
    if (tid < HDIM) fc_lds[tid] = fc_w[tid];

    // Loop-invariant B fragments: w_hh^T (K=96 x N=384) as f16, WMMA B layout.
    // Assumed 16-bit B layout (mirror of documented A layout, M<->N roles):
    //   lane n  (0..15): elems 0..7 = B[K=0..7][n],  8..15 = B[K=16..23][n]
    //   lane n+16      : elems 0..7 = B[K=8..15][n], 8..15 = B[K=24..31][n]
    // B[k][n] == w_hh[n*H + k] (contiguous in k -> coalesced-ish one-time load).
    v16h bfrag[TPW][KFRAGS];
    {
        const int nl   = lane & 15;
        const int koff = (lane >> 4) * 8;
        for (int ti = 0; ti < TPW; ++ti) {
            const int n = 16 * (TPW * wave + ti) + nl;
            for (int kf = 0; kf < KFRAGS; ++kf) {
                const int kb = 32 * kf + koff;
                v16h f;
                #pragma unroll
                for (int e = 0; e < 8; ++e) {
                    f[e]     = (_Float16)w_hh[n * HDIM + kb + e];
                    f[e + 8] = (_Float16)w_hh[n * HDIM + kb + 16 + e];
                }
                bfrag[ti][kf] = f;
            }
        }
    }

    // Per-thread elementwise constants (loop-invariant) and cell state in regs.
    int   eb[EPT], ehh[EPT];
    float wii[EPT], wif[EPT], wig[EPT], wio[EPT];
    float bbi[EPT], bbf[EPT], bbg[EPT], bbo[EPT];
    float cst[EPT];
    #pragma unroll
    for (int j = 0; j < EPT; ++j) {
        const int idx = tid + NTHREADS * j;   // 0..1535 over 16x96 state
        eb[j]  = idx / HDIM;
        ehh[j] = idx % HDIM;
        const int hh = ehh[j];
        wii[j] = w_ih[0 * HDIM + hh]; bbi[j] = b_ih[0 * HDIM + hh] + b_hh[0 * HDIM + hh];
        wif[j] = w_ih[1 * HDIM + hh]; bbf[j] = b_ih[1 * HDIM + hh] + b_hh[1 * HDIM + hh];
        wig[j] = w_ih[2 * HDIM + hh]; bbg[j] = b_ih[2 * HDIM + hh] + b_hh[2 * HDIM + hh];
        wio[j] = w_ih[3 * HDIM + hh]; bbo[j] = b_ih[3 * HDIM + hh] + b_hh[3 * HDIM + hh];
        cst[j] = 0.f;
    }
    const float fcb = fc_b[0];

    __syncthreads();

    for (int t = 0; t < T_STEPS; ++t) {
        // ---- stage next 64 timesteps of x into LDS ----
        if ((t & (CHUNK - 1)) == 0) {
            __syncthreads();   // prior step's readers of x_lds are done
            const float* xg = x + (size_t)t * BATCH + b0;
            #pragma unroll
            for (int j = 0; j < (CHUNK * BPB) / NTHREADS; ++j) {  // 4
                const int flat = tid + NTHREADS * j;
                x_lds[flat] = xg[(flat >> 4) * BATCH + (flat & 15)];
            }
            if (t + CHUNK < T_STEPS)
                __builtin_prefetch(x + (size_t)(t + CHUNK) * BATCH + b0, 0, 1);
            __syncthreads();
        }

        // ---- WMMA phase: g = h @ w_hh^T ----
        {
            const int m    = lane & 15;
            const int koff = (lane >> 4) * 8;
            v16h afrag[KFRAGS];
            #pragma unroll
            for (int kf = 0; kf < KFRAGS; ++kf) {
                // A layout (16-bit, 16x32): lane<16 holds K {0..7,16..23},
                // lane>=16 holds K {8..15,24..31}; two 16B ds loads each.
                const _Float16* p = &h16[m * HS + 32 * kf + koff];
                v8h lo = *(const v8h*)(p);
                v8h hi = *(const v8h*)(p + 16);
                afrag[kf] = __builtin_shufflevector(lo, hi,
                    0, 1, 2, 3, 4, 5, 6, 7, 8, 9, 10, 11, 12, 13, 14, 15);
            }
            const int row8 = 8 * (lane >> 4);
            #pragma unroll
            for (int ti = 0; ti < TPW; ++ti) {
                v8f acc = {};
                #pragma unroll
                for (int kf = 0; kf < KFRAGS; ++kf) {
                    acc = __builtin_amdgcn_wmma_f32_16x16x32_f16(
                        /*neg_a=*/false, afrag[kf],
                        /*neg_b=*/false, bfrag[ti][kf],
                        /*c_mod=*/(short)0, acc,
                        /*reuse_a=*/false, /*reuse_b=*/false);
                }
                // D layout: elem r -> row (r + 8*(lane>>4)), col (lane&15)
                const int col = 16 * (TPW * wave + ti) + (lane & 15);
                #pragma unroll
                for (int r = 0; r < 8; ++r)
                    g_lds[(row8 + r) * GS + col] = acc[r];
            }
        }
        __syncthreads();   // gates complete

        // ---- elementwise: gates_x add, activations, c/h update ----
        #pragma unroll
        for (int j = 0; j < EPT; ++j) {
            const int   b  = eb[j];
            const int   hh = ehh[j];
            const float xb = x_lds[(t & (CHUNK - 1)) * BPB + b];
            const float gi = g_lds[b * GS +   0 + hh] + __builtin_fmaf(xb, wii[j], bbi[j]);
            const float gf = g_lds[b * GS +  96 + hh] + __builtin_fmaf(xb, wif[j], bbf[j]);
            const float gg = g_lds[b * GS + 192 + hh] + __builtin_fmaf(xb, wig[j], bbg[j]);
            const float go = g_lds[b * GS + 288 + hh] + __builtin_fmaf(xb, wio[j], bbo[j]);
            const float iv = sigmoid_fast(gi);
            const float fv = sigmoid_fast(gf);
            const float gv = tanh_fast(gg);
            const float ov = sigmoid_fast(go);
            const float c  = __builtin_fmaf(fv, cst[j], iv * gv);
            cst[j] = c;
            const float h = ov * tanh_fast(c);
            h16[b * HS  + hh] = (_Float16)h;
            h32[b * H32S + hh] = h;
        }
        __syncthreads();   // h ready for output dot and next step's WMMA

        // ---- output: out[t,b] = h.fc_w + fc_b + x[t,b] (2 rows per wave) ----
        {
            const int drow = 2 * wave + (lane >> 4);
            const int k0   = lane & 15;
            float acc = 0.f;
            #pragma unroll
            for (int j = 0; j < HDIM / 16; ++j)
                acc = __builtin_fmaf(h32[drow * H32S + k0 + 16 * j],
                                     fc_lds[k0 + 16 * j], acc);
            #pragma unroll
            for (int off = 8; off > 0; off >>= 1)
                acc += __shfl_xor(acc, off, 32);
            if (k0 == 0)
                out[(size_t)t * BATCH + b0 + drow] =
                    acc + fcb + x_lds[(t & (CHUNK - 1)) * BPB + drow];
        }
        // No barrier needed here: next step's writes to g_lds/h16 are already
        // separated from this step's readers by the two barriers above.
    }
}

extern "C" void kernel_launch(void* const* d_in, const int* in_sizes, int n_in,
                              void* d_out, int out_size, void* d_ws, size_t ws_size,
                              hipStream_t stream) {
    const float* x    = (const float*)d_in[0];
    const float* w_ih = (const float*)d_in[1];
    const float* w_hh = (const float*)d_in[2];
    const float* b_ih = (const float*)d_in[3];
    const float* b_hh = (const float*)d_in[4];
    const float* fc_w = (const float*)d_in[5];
    const float* fc_b = (const float*)d_in[6];
    float* out = (float*)d_out;
    (void)in_sizes; (void)n_in; (void)out_size; (void)d_ws; (void)ws_size;

    lstm_persistent_kernel<<<dim3(BATCH / BPB), dim3(NTHREADS), 0, stream>>>(
        x, w_ih, w_hh, b_ih, b_hh, fc_w, fc_b, out);
}
